// MHA_30958124270314
// MI455X (gfx1250) — matrix operand order repaired and checked by reading, not compiled, not run on hardware
//
#include <hip/hip_runtime.h>
#include <stdint.h>

#define N_B   4
#define N_SEQ 2048
#define N_DIM 256
#define N_H   8
#define N_C   64
#define N_HC  512
#define N_OUT 256

typedef __attribute__((ext_vector_type(16))) __bf16 bf16x16;
typedef __attribute__((ext_vector_type(8)))  float  f32x8;

union Frag { uint32_t u[8]; bf16x16 v; };

__device__ __forceinline__ uint16_t f2bf(float f) {
    union { float f; uint32_t u; } x; x.f = f;
    uint32_t u = x.u;
    u += 0x7fffu + ((u >> 16) & 1u);   // round-to-nearest-even
    return (uint16_t)(u >> 16);
}

// A-matrix (16x32 bf16) per-lane K base for vgpr v, lane-half h
__device__ __forceinline__ int a_k0(int v, int half) {
    return ((v >> 2) << 4) + half * 8 + ((v & 3) << 1);
}
// B-matrix (32x16 bf16) per-lane K base for vgpr v, lane-half h
__device__ __forceinline__ int b_k0(int v, int half) {
    return ((v >> 2) << 4) + half * 8 + ((v & 3) << 1);
}

__device__ __forceinline__ f32x8 wmma_bf16(const Frag& a, const Frag& b, f32x8 c) {
    return __builtin_amdgcn_wmma_f32_16x16x32_bf16(false, a.v, false, b.v,
                                                   (short)0, c, false, false);
}

// ---------------------------------------------------------------------------
// Kernel 1: fp32 -> bf16 conversions (+ weight transposes for B-fragment reads)
// ---------------------------------------------------------------------------
__global__ __launch_bounds__(256)
void prep_kernel(const float* __restrict__ X,  const float* __restrict__ Wq,
                 const float* __restrict__ Wk, const float* __restrict__ Wv,
                 const float* __restrict__ Wo,
                 uint16_t* __restrict__ Xb,  uint16_t* __restrict__ WqT,
                 uint16_t* __restrict__ WkT, uint16_t* __restrict__ WvT,
                 uint16_t* __restrict__ WoT) {
    int i = blockIdx.x * blockDim.x + threadIdx.x;
    const int NX = N_B * N_SEQ * N_DIM;     // 2097152
    const int NW = N_H * N_DIM * N_C;       // 131072
    const int NO = N_HC * N_OUT;            // 131072
    if (i < NX) { Xb[i] = f2bf(X[i]); return; }
    i -= NX;
    if (i < 3 * NW) {
        int mat = i / NW, j = i % NW;
        int hh  = j >> 14;                  // / (64*256)
        int rem = j & 16383;
        int ch  = rem >> 8, dd = rem & 255; // WT[h][c][d] = W[h][d][c]
        const float* W = (mat == 0) ? Wq : (mat == 1) ? Wk : Wv;
        uint16_t*   WT = (mat == 0) ? WqT : (mat == 1) ? WkT : WvT;
        WT[j] = f2bf(W[((size_t)hh * N_DIM + dd) * N_C + ch]);
        return;
    }
    i -= 3 * NW;
    if (i < NO) {                           // WoT[n][k] = Wo[k][n]
        int n = i >> 9, kk = i & 511;
        WoT[i] = f2bf(Wo[(size_t)kk * N_OUT + n]);
    }
}

// ---------------------------------------------------------------------------
// Kernel 2: QKV projection.  Q,K -> [b,h,N,c] bf16 (Q pre-scaled 1/sqrt(c)),
//           V -> transposed [b,h,c,N] bf16.
// ---------------------------------------------------------------------------
__global__ __launch_bounds__(256)
void qkv_kernel(const uint16_t* __restrict__ Xb,
                const uint16_t* __restrict__ WqT, const uint16_t* __restrict__ WkT,
                const uint16_t* __restrict__ WvT,
                uint16_t* __restrict__ Qg, uint16_t* __restrict__ Kg,
                uint16_t* __restrict__ VTg) {
    int idx = blockIdx.x;
    int rt  = idx & 15;  idx >>= 4;         // row tile (128 rows each)
    int h   = idx & 7;   idx >>= 3;
    int b   = idx & 3;   idx >>= 2;
    int mat = idx;                          // 0=Q 1=K 2=V
    int lane = threadIdx.x & 31, w = threadIdx.x >> 5;
    int m = lane & 15, half = lane >> 4;

    const uint16_t* WT = (mat == 0) ? WqT : (mat == 1) ? WkT : WvT;
    const uint16_t* Xrow = Xb + ((size_t)b * N_SEQ + rt * 128 + w * 16 + m) * N_DIM;

    f32x8 acc[4] = {};
    for (int kc = 0; kc < 8; ++kc) {
        Frag a;
#pragma unroll
        for (int v = 0; v < 8; ++v)
            a.u[v] = *(const uint32_t*)(Xrow + kc * 32 + a_k0(v, half));
#pragma unroll
        for (int t = 0; t < 4; ++t) {
            Frag bb;
            const uint16_t* Wc = WT + ((size_t)h * N_C + t * 16 + m) * N_DIM + kc * 32;
#pragma unroll
            for (int v = 0; v < 8; ++v)
                bb.u[v] = *(const uint32_t*)(Wc + b_k0(v, half));
            acc[t] = wmma_bf16(a, bb, acc[t]);
        }
    }
    float scale = (mat == 0) ? 0.125f : 1.0f;  // 1/sqrt(64) folded into Q
    int row0 = rt * 128 + w * 16 + 8 * half;   // vgpr v -> row0 + v
    if (mat < 2) {
        uint16_t* Outp = (mat == 0 ? Qg : Kg) + ((size_t)b * N_H + h) * N_SEQ * N_C;
#pragma unroll
        for (int t = 0; t < 4; ++t)
#pragma unroll
            for (int v = 0; v < 8; ++v)
                Outp[(size_t)(row0 + v) * N_C + t * 16 + m] = f2bf(acc[t][v] * scale);
    } else {
        uint16_t* Outp = VTg + ((size_t)b * N_H + h) * N_C * N_SEQ;
#pragma unroll
        for (int t = 0; t < 4; ++t)
#pragma unroll
            for (int v = 0; v < 8; ++v)
                Outp[(size_t)(t * 16 + m) * N_SEQ + row0 + v] = f2bf(acc[t][v]);
    }
}

// ---------------------------------------------------------------------------
// Kernel 3: flash attention.  Each wave owns 16 queries; 32-key blocks.
// Output written concat-head layout vc[b][n][h*c] in bf16.
// ---------------------------------------------------------------------------
__global__ __launch_bounds__(256)
void attn_kernel(const uint16_t* __restrict__ Qg, const uint16_t* __restrict__ Kg,
                 const uint16_t* __restrict__ VTg, uint16_t* __restrict__ vc) {
    __shared__ uint16_t lds_p[8][16][32];   // per-wave P tile (C-layout -> A-layout)
    int idx = blockIdx.x;
    int qt = idx & 15;
    int h  = (idx >> 4) & 7;
    int b  = idx >> 7;
    int lane = threadIdx.x & 31, w = threadIdx.x >> 5;
    int m = lane & 15, half = lane >> 4;
    size_t bh = (size_t)b * N_H + h;
    int r0 = qt * 128 + w * 16;

    // Q A-fragments (16x64 -> two 16x32 frags), Q is pre-scaled
    const uint16_t* Qrow = Qg + (bh * N_SEQ + r0 + m) * N_C;
    Frag qa[2];
#pragma unroll
    for (int kc = 0; kc < 2; ++kc)
#pragma unroll
        for (int v = 0; v < 8; ++v)
            qa[kc].u[v] = *(const uint32_t*)(Qrow + kc * 32 + a_k0(v, half));

    float mrun[8], lrun[8];
    f32x8 oacc[4] = {};
#pragma unroll
    for (int v = 0; v < 8; ++v) { mrun[v] = -1e30f; lrun[v] = 0.0f; }

    const uint16_t* Kbase = Kg + bh * (size_t)N_SEQ * N_C;
    const uint16_t* Vbase = VTg + bh * (size_t)N_C * N_SEQ;

    for (int kb = 0; kb < N_SEQ / 32; ++kb) {
        int key0 = kb * 32;
        // S = Q K^T : two 16x16 tiles covering 32 keys
        f32x8 s[2] = {};
#pragma unroll
        for (int st = 0; st < 2; ++st) {
            const uint16_t* Krow = Kbase + (size_t)(key0 + st * 16 + m) * N_C;
#pragma unroll
            for (int kc = 0; kc < 2; ++kc) {
                Frag bb;
#pragma unroll
                for (int v = 0; v < 8; ++v)
                    bb.u[v] = *(const uint32_t*)(Krow + kc * 32 + b_k0(v, half));
                s[st] = wmma_bf16(qa[kc], bb, s[st]);
            }
        }
        // Online softmax (row = v + 8*half lives across 16 lanes of this half)
#pragma unroll
        for (int v = 0; v < 8; ++v) {
            float rmax = fmaxf(s[0][v], s[1][v]);
#pragma unroll
            for (int off = 1; off < 16; off <<= 1)
                rmax = fmaxf(rmax, __shfl_xor(rmax, off, 32));
            float mnew = fmaxf(mrun[v], rmax);
            float scl  = __expf(mrun[v] - mnew);
            float p0 = __expf(s[0][v] - mnew);
            float p1 = __expf(s[1][v] - mnew);
            float rs = p0 + p1;
#pragma unroll
            for (int off = 1; off < 16; off <<= 1)
                rs += __shfl_xor(rs, off, 32);
            lrun[v] = lrun[v] * scl + rs;
            mrun[v] = mnew;
#pragma unroll
            for (int t = 0; t < 4; ++t) oacc[t][v] *= scl;
            int prow = v + 8 * half;
            lds_p[w][prow][m]      = f2bf(p0);
            lds_p[w][prow][16 + m] = f2bf(p1);
        }
        __syncthreads();   // LDS write->read within wave (uniform loop; cheap)
        Frag pa;
#pragma unroll
        for (int v = 0; v < 8; ++v)
            pa.u[v] = *(const uint32_t*)(&lds_p[w][m][a_k0(v, half)]);
        // O += P V  (V transposed -> contiguous pairs)
#pragma unroll
        for (int t = 0; t < 4; ++t) {
            Frag bb;
            const uint16_t* Vcol = Vbase + (size_t)(t * 16 + m) * N_SEQ + key0;
#pragma unroll
            for (int v = 0; v < 8; ++v)
                bb.u[v] = *(const uint32_t*)(Vcol + b_k0(v, half));
            oacc[t] = wmma_bf16(pa, bb, oacc[t]);
        }
        __syncthreads();
    }
    // Normalize and write concat-head vc[b][row][h*64 + ch] (bf16)
#pragma unroll
    for (int v = 0; v < 8; ++v) {
        float inv = 1.0f / lrun[v];
        int row = r0 + v + 8 * half;
        uint16_t* dst = vc + ((size_t)b * N_SEQ + row) * N_HC + h * N_C;
#pragma unroll
        for (int t = 0; t < 4; ++t)
            dst[t * 16 + m] = f2bf(oacc[t][v] * inv);
    }
}

// ---------------------------------------------------------------------------
// Kernel 4: output projection  [8192,512]bf16 x [512,256] -> fp32 out
// ---------------------------------------------------------------------------
__global__ __launch_bounds__(256)
void oproj_kernel(const uint16_t* __restrict__ vc, const uint16_t* __restrict__ WoT,
                  float* __restrict__ out) {
    int idx = blockIdx.x;
    int ct = idx & 3;            // 64-col tile
    int rtile = idx >> 2;        // 128-row tile (64 total)
    int lane = threadIdx.x & 31, w = threadIdx.x >> 5;
    int m = lane & 15, half = lane >> 4;
    int r0 = rtile * 128 + w * 16;

    const uint16_t* Arow = vc + (size_t)(r0 + m) * N_HC;
    f32x8 acc[4] = {};
    for (int kc = 0; kc < N_HC / 32; ++kc) {
        Frag a;
#pragma unroll
        for (int v = 0; v < 8; ++v)
            a.u[v] = *(const uint32_t*)(Arow + kc * 32 + a_k0(v, half));
#pragma unroll
        for (int t = 0; t < 4; ++t) {
            Frag bb;
            const uint16_t* Wc = WoT + (size_t)(ct * 64 + t * 16 + m) * N_HC + kc * 32;
#pragma unroll
            for (int v = 0; v < 8; ++v)
                bb.u[v] = *(const uint32_t*)(Wc + b_k0(v, half));
            acc[t] = wmma_bf16(a, bb, acc[t]);
        }
    }
    int rowbase = r0 + 8 * half;
#pragma unroll
    for (int t = 0; t < 4; ++t)
#pragma unroll
        for (int v = 0; v < 8; ++v)
            out[(size_t)(rowbase + v) * N_OUT + ct * 64 + t * 16 + m] = acc[t][v];
}

// ---------------------------------------------------------------------------
extern "C" void kernel_launch(void* const* d_in, const int* in_sizes, int n_in,
                              void* d_out, int out_size, void* d_ws, size_t ws_size,
                              hipStream_t stream) {
    (void)in_sizes; (void)n_in; (void)out_size; (void)ws_size;
    const float* X  = (const float*)d_in[0];
    const float* Wq = (const float*)d_in[1];
    const float* Wk = (const float*)d_in[2];
    const float* Wv = (const float*)d_in[3];
    const float* Wo = (const float*)d_in[4];
    float* out = (float*)d_out;

    uint8_t* ws = (uint8_t*)d_ws;
    const size_t SZ_X  = (size_t)N_B * N_SEQ * N_DIM * 2;   // 4 MiB
    const size_t SZ_W  = (size_t)N_H * N_DIM * N_C * 2;     // 256 KiB each
    const size_t SZ_WO = (size_t)N_HC * N_OUT * 2;          // 256 KiB
    const size_t SZ_Q  = (size_t)N_B * N_H * N_SEQ * N_C * 2; // 8 MiB

    uint16_t* Xb  = (uint16_t*)(ws);
    uint16_t* WqT = (uint16_t*)(ws + SZ_X);
    uint16_t* WkT = (uint16_t*)(ws + SZ_X + SZ_W);
    uint16_t* WvT = (uint16_t*)(ws + SZ_X + 2 * SZ_W);
    uint16_t* WoT = (uint16_t*)(ws + SZ_X + 3 * SZ_W);
    uint16_t* Qg  = (uint16_t*)(ws + SZ_X + 3 * SZ_W + SZ_WO);
    uint16_t* Kg  = (uint16_t*)((uint8_t*)Qg + SZ_Q);
    uint16_t* VTg = (uint16_t*)((uint8_t*)Kg + SZ_Q);
    uint16_t* vcb = (uint16_t*)((uint8_t*)VTg + SZ_Q);

    const int NTOT = N_B * N_SEQ * N_DIM + 3 * N_H * N_DIM * N_C + N_HC * N_OUT;
    prep_kernel<<<(NTOT + 255) / 256, 256, 0, stream>>>(X, Wq, Wk, Wv, Wo,
                                                        Xb, WqT, WkT, WvT, WoT);
    qkv_kernel<<<3 * N_B * N_H * (N_SEQ / 128), 256, 0, stream>>>(Xb, WqT, WkT, WvT,
                                                                  Qg, Kg, VTg);
    attn_kernel<<<N_B * N_H * (N_SEQ / 128), 256, 0, stream>>>(Qg, Kg, VTg, vcb);
    oproj_kernel<<<(N_B * N_SEQ / 128) * (N_OUT / 64), 256, 0, stream>>>(vcb, WoT, out);
}
